// PACA_73521250173707
// MI455X (gfx1250) — 16-bit and 8-bit matrix operand order repaired and checked by reading, compiled, not run on hardware
//
#include <hip/hip_runtime.h>
#include <math.h>

typedef _Float16 half_t;
typedef unsigned char u8;
typedef __attribute__((ext_vector_type(16))) _Float16 v16h;
typedef __attribute__((ext_vector_type(8)))  _Float16 v8h;
typedef __attribute__((ext_vector_type(4)))  _Float16 v4h;
typedef __attribute__((ext_vector_type(8)))  float    v8f;
typedef __attribute__((ext_vector_type(4)))  float    v4f;
typedef __attribute__((ext_vector_type(2)))  int      v2i;
typedef __attribute__((ext_vector_type(4)))  int      v4i;
typedef __attribute__((ext_vector_type(8)))  int      v8i;
typedef __attribute__((ext_vector_type(16))) int      v16i;

#define BATCH 4
#define CCH   256
#define NSP   4096

__device__ __forceinline__ v16h load_a_frag(const half_t* buf, int rs, int row0, int k0) {
  const int lane = threadIdx.x & 31;
  const int m = lane & 15, hi = lane >> 4;
  const half_t* p = buf + (size_t)(row0 + m) * rs + k0 + 8 * hi;
  v8h lo = *(const v8h*)(p);
  v8h hv = *(const v8h*)(p + 16);
  v16h r;
#pragma unroll
  for (int i = 0; i < 8; ++i) { r[i] = lo[i]; r[i + 8] = hv[i]; }
  return r;
}
__device__ __forceinline__ v16h load_b_frag(const half_t* buf, int rs, int row0, int k0) {
  const int lane = threadIdx.x & 31;
  const int n = lane & 15, hi = lane >> 4;
  const half_t* p = buf + (size_t)(row0 + n) * rs + k0 + 8 * hi;
  v8h lo = *(const v8h*)(p);
  v8h hv = *(const v8h*)(p + 16);
  v16h r;
#pragma unroll
  for (int i = 0; i < 8; ++i) { r[i] = lo[i]; r[i + 8] = hv[i]; }
  return r;
}
__device__ __forceinline__ v8f wmma_f16(v16h a, v16h b, v8f c) {
  return __builtin_amdgcn_wmma_f32_16x16x32_f16(false, a, false, b, (short)0, c,
                                                false, false);
}

__device__ __forceinline__ v8i load_a8_64(const u8* buf, int rs, int row0, int k0) {
  const int lane = threadIdx.x & 31;
  const int m = lane & 15, hi = lane >> 4;
  const u8* p = buf + (size_t)(row0 + m) * rs + k0 + 16 * hi;
  v2i c0 = *(const v2i*)(p);
  v2i c1 = *(const v2i*)(p + 8);
  v2i c2 = *(const v2i*)(p + 32);
  v2i c3 = *(const v2i*)(p + 40);
  return (v8i){c0[0], c0[1], c1[0], c1[1], c2[0], c2[1], c3[0], c3[1]};
}
__device__ __forceinline__ v16i load_a8_128(const u8* buf, int rs, int row0, int k0) {
  const int lane = threadIdx.x & 31;
  const int m = lane & 15, hi = lane >> 4;
  const u8* p = buf + (size_t)(row0 + m) * rs + k0 + 16 * hi;
  v16i r;
#pragma unroll
  for (int g = 0; g < 4; ++g) {
    v2i lo = *(const v2i*)(p + (g >> 1) * 32 + (g & 1) * 8);
    v2i hv = *(const v2i*)(p + 64 + (g >> 1) * 32 + (g & 1) * 8);
    r[2 * g]     = lo[0]; r[2 * g + 1] = lo[1];
    r[8 + 2 * g] = hv[0]; r[9 + 2 * g] = hv[1];
  }
  return r;
}
__device__ __forceinline__ v8i load_b8_64(const u8* buf, int rs, int row0, int k0) {
  const int lane = threadIdx.x & 31;
  const int n = lane & 15, hi = lane >> 4;
  const u8* p = buf + (size_t)(row0 + n) * rs + k0 + 16 * hi;
  v4i c0 = *(const v4i*)(p);
  v4i c1 = *(const v4i*)(p + 32);
  return (v8i){c0[0], c0[1], c0[2], c0[3], c1[0], c1[1], c1[2], c1[3]};
}
__device__ __forceinline__ v16i load_b8_128(const u8* buf, int rs, int row0, int k0) {
  const int lane = threadIdx.x & 31;
  const int n = lane & 15, hi = lane >> 4;
  const u8* p = buf + (size_t)(row0 + n) * rs + k0 + 16 * hi;
  v16i r;
#pragma unroll
  for (int g = 0; g < 4; ++g) {
    v4i c = *(const v4i*)(p + g * 32);
    r[4 * g] = c[0]; r[4 * g + 1] = c[1]; r[4 * g + 2] = c[2]; r[4 * g + 3] = c[3];
  }
  return r;
}
__device__ __forceinline__ v8f wmma_fp8_128(v16i a, v16i b, v8f c) {
  return __builtin_amdgcn_wmma_f32_16x16x128_fp8_fp8(a, b, (short)0, c, false, false);
}
__device__ __forceinline__ v8f wmma_fp8_64(v8i a, v8i b, v8f c) {
  return __builtin_amdgcn_wmma_f32_16x16x64_fp8_fp8(a, b, (short)0, c, false, false);
}

__device__ __forceinline__ u8 f32_fp8(float x) {
#if __has_builtin(__builtin_amdgcn_cvt_pk_fp8_f32)
  return (u8)(__builtin_amdgcn_cvt_pk_fp8_f32(x, x, 0, false) & 0xff);
#else
  unsigned u = __float_as_uint(x);
  unsigned s = (u >> 24) & 0x80u;
  unsigned a = u & 0x7fffffffu;
  if (a > 0x43e00000u) a = 0x43e00000u;
  if (a < 0x3c000000u) {
    float q = __uint_as_float(a) * 512.0f + 0.5f;
    return (u8)(s | ((unsigned)q & 7u));
  }
  a += 0x000FFFFFu + ((a >> 20) & 1u);
  if (a > 0x43e00000u) a = 0x43e00000u;
  return (u8)(s | ((((a >> 23) - 120u) << 3) & 0x78u) | ((a >> 20) & 7u));
#endif
}

__device__ __forceinline__ float redmax16(float v) {
  v = fmaxf(v, __shfl_xor(v, 1, 32));
  v = fmaxf(v, __shfl_xor(v, 2, 32));
  v = fmaxf(v, __shfl_xor(v, 4, 32));
  v = fmaxf(v, __shfl_xor(v, 8, 32));
  return v;
}
__device__ __forceinline__ float redsum16(float v) {
  v += __shfl_xor(v, 1, 32);
  v += __shfl_xor(v, 2, 32);
  v += __shfl_xor(v, 4, 32);
  v += __shfl_xor(v, 8, 32);
  return v;
}

__device__ __forceinline__ void async_b128(u8* lds, const u8* g) {
  asm volatile("global_load_async_to_lds_b128 %0, %1, off"
               :
               : "v"((unsigned)(unsigned long long)lds),
                 "v"((unsigned long long)g)
               : "memory");
}
__device__ __forceinline__ void wait_async0() {
#if __has_builtin(__builtin_amdgcn_s_wait_asynccnt)
  __builtin_amdgcn_s_wait_asynccnt(0);
#else
  asm volatile("s_wait_asynccnt 0x0" ::: "memory");
#endif
}

#define PM 128
#define PR 264

__global__ __launch_bounds__(256) void proj_kernel(
    const float* __restrict__ src, const float* __restrict__ Wt,
    const float* __restrict__ bias, u8* __restrict__ dst, int vtrans) {
  __shared__ half_t As[PM * PR];
  __shared__ half_t Ws[128 * PR];

  const int tid  = threadIdx.x;
  const int wave = tid >> 5;
  const int lane = tid & 31;
  const int hi = lane >> 4, ln = lane & 15;
  const int m0 = blockIdx.x * PM;
  const int b  = blockIdx.y;
  const int mw = wave * 16;

  for (int idx = tid; idx < CCH * (PM / 4); idx += 256) {
    int c  = idx / (PM / 4);
    int m4 = (idx % (PM / 4)) * 4;
    v4f v = *(const v4f*)(src + ((size_t)b * CCH + c) * NSP + m0 + m4);
#pragma unroll
    for (int j = 0; j < 4; ++j) As[(size_t)(m4 + j) * PR + c] = (half_t)v[j];
  }

  for (int dh = 0; dh < 2; ++dh) {
    __syncthreads();
    for (int idx = tid; idx < 128 * (CCH / 4); idx += 256) {
      int dl = idx / (CCH / 4);
      int c4 = (idx % (CCH / 4)) * 4;
      v4f v = *(const v4f*)(Wt + (size_t)(dh * 128 + dl) * CCH + c4);
      v4h h;
#pragma unroll
      for (int j = 0; j < 4; ++j) h[j] = (half_t)v[j];
      *(v4h*)(Ws + (size_t)dl * PR + c4) = h;
    }
    __syncthreads();

    v8f acc[8];
#pragma unroll
    for (int i = 0; i < 8; ++i) acc[i] = (v8f){0.f,0.f,0.f,0.f,0.f,0.f,0.f,0.f};

#pragma unroll
    for (int kb = 0; kb < 8; ++kb) {
      v16h a = load_a_frag(As, PR, mw, kb * 32);
#pragma unroll
      for (int d4 = 0; d4 < 8; d4 += 4) {
        v16h b0 = load_b_frag(Ws, PR, (d4 + 0) * 16, kb * 32);
        v16h b1 = load_b_frag(Ws, PR, (d4 + 1) * 16, kb * 32);
        v16h b2 = load_b_frag(Ws, PR, (d4 + 2) * 16, kb * 32);
        v16h b3 = load_b_frag(Ws, PR, (d4 + 3) * 16, kb * 32);
        acc[d4 + 0] = wmma_f16(a, b0, acc[d4 + 0]);
        acc[d4 + 1] = wmma_f16(a, b1, acc[d4 + 1]);
        acc[d4 + 2] = wmma_f16(a, b2, acc[d4 + 2]);
        acc[d4 + 3] = wmma_f16(a, b3, acc[d4 + 3]);
      }
    }

#pragma unroll
    for (int db = 0; db < 8; ++db) {
      int d = dh * 128 + db * 16 + ln;
      float bv = bias[d];
#pragma unroll
      for (int r = 0; r < 8; ++r) {
        int m = m0 + mw + r + 8 * hi;
        u8 val = f32_fp8(acc[db][r] + bv);
        if (vtrans)
          dst[((size_t)b * CCH + d) * NSP + m] = val;
        else
          dst[((size_t)b * NSP + m) * CCH + d] = val;
      }
    }
  }
}

#define AM   128
#define KT   64
#define QR8  272
#define VR8  80
#define OBR  65

__global__ __launch_bounds__(256) void attn_kernel(
    const u8* __restrict__ Qg,
    const u8* __restrict__ Kg,
    const u8* __restrict__ Vtg,
    const float* __restrict__ xres,
    float* __restrict__ out) {
  __shared__ __align__(16) unsigned char smemQ[AM * QR8];
  __shared__ u8 Ks8[2][KT * QR8];
  __shared__ u8 Vts8[2][CCH * VR8];
  __shared__ u8 Ps8[8][16 * VR8];
  u8*    Qs8  = (u8*)smemQ;
  float* Obuf = (float*)smemQ;

  const int tid  = threadIdx.x;
  const int wave = tid >> 5;
  const int lane = tid & 31;
  const int hi = lane >> 4, ln = lane & 15;
  const int q0 = blockIdx.x * AM;
  const int b  = blockIdx.y;
  const int mw = wave * 16;
  const float scale = 0.0625f;

  for (int idx = tid; idx < AM * 16; idx += 256) {
    int row = idx >> 4, c16 = (idx & 15) * 16;
    *(v4i*)(Qs8 + (size_t)row * QR8 + c16) =
        *(const v4i*)(Qg + ((size_t)b * NSP + q0 + row) * CCH + c16);
  }

  auto prefetch = [&](int buf, int kt) {
    {
      const u8* g = Kg + ((size_t)b * NSP + kt + (tid >> 4)) * CCH + (tid & 15) * 16;
      u8* l = &Ks8[buf][(size_t)(tid >> 4) * QR8 + (tid & 15) * 16];
#pragma unroll
      for (int i = 0; i < 4; ++i)
        async_b128(l + i * 16 * QR8, g + i * 16 * CCH);
    }
    {
      const u8* g = Vtg + ((size_t)b * CCH + (tid >> 2)) * NSP + kt + (tid & 3) * 16;
      u8* l = &Vts8[buf][(size_t)(tid >> 2) * VR8 + (tid & 3) * 16];
#pragma unroll
      for (int i = 0; i < 4; ++i)
        async_b128(l + i * 64 * VR8, g + (size_t)i * 64 * NSP);
    }
  };

  prefetch(0, 0);

  v8f accO[16];
#pragma unroll
  for (int i = 0; i < 16; ++i) accO[i] = (v8f){0.f,0.f,0.f,0.f,0.f,0.f,0.f,0.f};
  float rm[8], rl[8];
#pragma unroll
  for (int r = 0; r < 8; ++r) { rm[r] = -INFINITY; rl[r] = 0.f; }

  for (int it = 0; it < NSP / KT; ++it) {
    const int cur = it & 1;
    wait_async0();
    __syncthreads();
    if (it + 1 < NSP / KT) prefetch(cur ^ 1, (it + 1) * KT);

    const u8* Kc = Ks8[cur];
    const u8* Vc = Vts8[cur];

    v8f s[4];
#pragma unroll
    for (int nb = 0; nb < 4; ++nb) s[nb] = (v8f){0.f,0.f,0.f,0.f,0.f,0.f,0.f,0.f};
#pragma unroll
    for (int kb = 0; kb < 2; ++kb) {
      v16i a = load_a8_128(Qs8, QR8, mw, kb * 128);
#pragma unroll
      for (int nb = 0; nb < 4; nb += 2) {
        v16i b0 = load_b8_128(Kc, QR8, nb * 16,      kb * 128);
        v16i b1 = load_b8_128(Kc, QR8, nb * 16 + 16, kb * 128);
        s[nb]     = wmma_fp8_128(a, b0, s[nb]);
        s[nb + 1] = wmma_fp8_128(a, b1, s[nb + 1]);
      }
    }

    float corr[8];
#pragma unroll
    for (int r = 0; r < 8; ++r) {
      float s0 = s[0][r] * scale, s1 = s[1][r] * scale;
      float s2 = s[2][r] * scale, s3 = s[3][r] * scale;
      float mx = redmax16(fmaxf(fmaxf(s0, s1), fmaxf(s2, s3)));
      float mnew = fmaxf(rm[r], mx);
      corr[r] = __expf(rm[r] - mnew);
      rm[r] = mnew;
      float p0 = __expf(s0 - mnew), p1 = __expf(s1 - mnew);
      float p2 = __expf(s2 - mnew), p3 = __expf(s3 - mnew);
      u8* prow = &Ps8[wave][(size_t)(r + 8 * hi) * VR8 + ln];
      prow[0]  = f32_fp8(p0);  prow[16] = f32_fp8(p1);
      prow[32] = f32_fp8(p2);  prow[48] = f32_fp8(p3);
      rl[r] = rl[r] * corr[r] + redsum16(p0 + p1 + p2 + p3);
    }
#pragma unroll
    for (int f = 0; f < 16; ++f)
#pragma unroll
      for (int r = 0; r < 8; ++r) accO[f][r] *= corr[r];

    v8i a = load_a8_64(&Ps8[wave][0], VR8, 0, 0);
#pragma unroll
    for (int f = 0; f < 16; f += 4) {
      v8i b0 = load_b8_64(Vc, VR8, (f + 0) * 16, 0);
      v8i b1 = load_b8_64(Vc, VR8, (f + 1) * 16, 0);
      v8i b2 = load_b8_64(Vc, VR8, (f + 2) * 16, 0);
      v8i b3 = load_b8_64(Vc, VR8, (f + 3) * 16, 0);
      accO[f + 0] = wmma_fp8_64(a, b0, accO[f + 0]);
      accO[f + 1] = wmma_fp8_64(a, b1, accO[f + 1]);
      accO[f + 2] = wmma_fp8_64(a, b2, accO[f + 2]);
      accO[f + 3] = wmma_fp8_64(a, b3, accO[f + 3]);
    }
    __syncthreads();
  }

  float inv[8];
#pragma unroll
  for (int r = 0; r < 8; ++r) inv[r] = 1.0f / rl[r];

  for (int p = 0; p < 4; ++p) {
    __syncthreads();
#pragma unroll
    for (int f = 0; f < 4; ++f) {
      int cf = p * 4 + f;
#pragma unroll
      for (int r = 0; r < 8; ++r)
        Obuf[(size_t)(mw + r + 8 * hi) * OBR + f * 16 + ln] = accO[cf][r] * inv[r];
    }
    __syncthreads();
    int cl = tid >> 2;
    int qh = (tid & 3) * 32;
    size_t gbase = ((size_t)b * CCH + p * 64 + cl) * NSP + q0 + qh;
#pragma unroll
    for (int j = 0; j < 32; ++j)
      out[gbase + j] = xres[gbase + j] + Obuf[(size_t)(qh + j) * OBR + cl];
  }
}

extern "C" void kernel_launch(void* const* d_in, const int* in_sizes, int n_in,
                              void* d_out, int out_size, void* d_ws, size_t ws_size,
                              hipStream_t stream) {
  const float* x  = (const float*)d_in[0];
  const float* y  = (const float*)d_in[1];
  const float* Wq = (const float*)d_in[2];
  const float* bq = (const float*)d_in[3];
  const float* Wk = (const float*)d_in[4];
  const float* bk = (const float*)d_in[5];
  const float* Wv = (const float*)d_in[6];
  const float* bv = (const float*)d_in[7];
  float* out = (float*)d_out;

  const size_t qkv = (size_t)BATCH * NSP * CCH;
  u8* Q8  = (u8*)d_ws;
  u8* K8  = Q8 + qkv;
  u8* Vt8 = K8 + qkv;

  dim3 gp(NSP / PM, BATCH);
  proj_kernel<<<gp, 256, 0, stream>>>(x, Wq, bq, Q8, 0);
  proj_kernel<<<gp, 256, 0, stream>>>(y, Wk, bk, K8, 0);
  proj_kernel<<<gp, 256, 0, stream>>>(y, Wv, bv, Vt8, 1);

  dim3 ga(NSP / AM, BATCH);
  attn_kernel<<<ga, 256, 0, stream>>>(Q8, K8, Vt8, x, out);
}
